// TriggerAwareGraphDQN_75024488726533
// MI455X (gfx1250) — matrix operand order repaired and checked by reading, not compiled, not run on hardware
//
#include <hip/hip_runtime.h>
#include <hip/hip_bf16.h>

#define NN 8192

typedef __attribute__((ext_vector_type(16))) _Float16 v16h;
typedef __attribute__((ext_vector_type(8)))  float    v8f;
typedef __attribute__((ext_vector_type(4)))  float    v4f;

union V16H {
  v16h v;
  float4 f4[2];
  unsigned u[8];
  _Float16 h[16];
};
union PK8 {
  uint4 u4;
  unsigned u[4];
  _Float16 h[8];
};

// ---------------------------------------------------------------------------
// Phase 0: scalar gate factor + stateful coefficient
// gate = mean(sigmoid(trig @ wg + bg)); c = stateful_bias * (1 - trig[0])
// ---------------------------------------------------------------------------
__global__ __launch_bounds__(64) void k_scalars(const float* __restrict__ trig,
                                                const float* __restrict__ wg,
                                                const float* __restrict__ bg,
                                                const float* __restrict__ sb,
                                                float* __restrict__ sc) {
  __shared__ float red[64];
  const int t = threadIdx.x;
  float g = fmaf(trig[0], wg[t], fmaf(trig[1], wg[64 + t], bg[t]));
  g = 1.0f / (1.0f + __expf(-g));
  red[t] = g;
  __syncthreads();
  for (int off = 32; off > 0; off >>= 1) {
    if (t < off) red[t] += red[t + off];
    __syncthreads();
  }
  if (t == 0) {
    sc[0] = red[0] * (1.0f / 64.0f);
    sc[1] = sb[0] * (1.0f - trig[0]);
  }
}

// ---------------------------------------------------------------------------
// Phase 1: per-row embed MLP.  2 threads per row; both compute full h, each
// emits half of q/k/m.  Writes h (f32), q/k (f16 row-major), mT (f16
// TRANSPOSED 64 x 8192, so the attention kernel's A-operand loads are
// contiguous), and s[i] = node_features[i,2].
// ---------------------------------------------------------------------------
__global__ __launch_bounds__(256) void k_embed(
    const float* __restrict__ nf, const float* __restrict__ trig,
    const float* __restrict__ w1, const float* __restrict__ b1,
    const float* __restrict__ w2, const float* __restrict__ b2,
    const float* __restrict__ wq, const float* __restrict__ bq,
    const float* __restrict__ wk, const float* __restrict__ bk,
    const float* __restrict__ wm, const float* __restrict__ bm,
    float* __restrict__ hOut, _Float16* __restrict__ qOut,
    _Float16* __restrict__ kOut, _Float16* __restrict__ mTOut,
    float* __restrict__ sOut) {
  const int gid  = blockIdx.x * 256 + threadIdx.x;  // 0 .. 16383
  const int row  = gid >> 1;
  const int half = gid & 1;

  const float x0 = nf[row * 3 + 0];
  const float x1 = nf[row * 3 + 1];
  const float x2 = nf[row * 3 + 2];
  const float x3 = trig[0];
  const float x4 = trig[1];

  float h1[64];
#pragma unroll
  for (int t = 0; t < 64; ++t) {
    float a = b1[t];
    a = fmaf(x0, w1[0 * 64 + t], a);
    a = fmaf(x1, w1[1 * 64 + t], a);
    a = fmaf(x2, w1[2 * 64 + t], a);
    a = fmaf(x3, w1[3 * 64 + t], a);
    a = fmaf(x4, w1[4 * 64 + t], a);
    h1[t] = fmaxf(a, 0.0f);
  }

  float hv[64];
#pragma unroll
  for (int t = 0; t < 64; ++t) {
    float a = b2[t];
#pragma unroll
    for (int c = 0; c < 64; ++c) a = fmaf(h1[c], w2[c * 64 + t], a);
    hv[t] = a;
  }

  const int t0 = half * 32;
#pragma unroll
  for (int e = 0; e < 32; ++e) {
    float v = (half == 0) ? hv[e] : hv[32 + e];   // keep hv constant-indexed
    hOut[(size_t)row * 64 + t0 + e] = v;
  }
  if (half == 0) sOut[row] = x2;

#pragma unroll 4
  for (int e = 0; e < 32; ++e) {
    const int t = t0 + e;
    float qa = bq[t], ka = bk[t], ma = bm[t];
#pragma unroll
    for (int c = 0; c < 64; ++c) {
      const float hc = hv[c];
      qa = fmaf(hc, wq[c * 64 + t], qa);
      ka = fmaf(hc, wk[c * 64 + t], ka);
      ma = fmaf(hc, wm[c * 64 + t], ma);
    }
    qOut[(size_t)row * 64 + t] = (_Float16)qa;
    kOut[(size_t)row * 64 + t] = (_Float16)ka;
    mTOut[(size_t)t * NN + row] = (_Float16)ma;   // lanes -> consecutive rows
  }
}

// ---------------------------------------------------------------------------
// Phase 2: fused attention, WMMA wave32.
// One 16-row block per workgroup; 4 waves split the j-range (stride 128).
// Per 16-col tile pair:
//   S'[j,i] = k_tile (A, M=j,K=d) x q_rows (B, K=d,N=i)  -> 2x WMMA f16 (K=64)
//   P = exp(gate/8 * S' + topo + c*s_i*s_j)  (0 where adj==0; no max needed,
//   scores are O(1) bounded so plain exp is exact)
//   rowsum += sum_j P   (per-lane partial, D layout has lane = i)
//   O^T  += mT_tile (A, M=d,K=j) x P' (B, K=j,N=i)       -> 4x WMMA f16
// D->B conversion of P' needs only a lane^16 half swap: ds_swizzle 0x401f.
// adj is 256 MB (> 192 MB L2) and strictly single-use -> non-temporal loads,
// so the L2 stays dedicated to the q/k/mT/s working set (~3 MB) that every
// row-block re-reads.  Partials reduced in LDS; z = h + O/rowsum.
// ---------------------------------------------------------------------------
__device__ __forceinline__ uint4 score_tile(const _Float16* __restrict__ kptr,
                                            const V16H& bq0, const V16H& bq1,
                                            const float* __restrict__ aptr,
                                            const float* __restrict__ sptr,
                                            float gate8, float csi, float& rs) {
  V16H ak0, ak1;
  const float4* kp = (const float4*)kptr;
  ak0.f4[0] = kp[0];   // d = base+0..7   (A layout: low lanes K 0..7 / hi 8..15)
  ak0.f4[1] = kp[2];   // d = base+16..23 (K 16..23 / 24..31)
  ak1.f4[0] = kp[4];   // +32
  ak1.f4[1] = kp[6];   // +48
  v8f sv;
#pragma unroll
  for (int e = 0; e < 8; ++e) sv[e] = 0.0f;
  sv = __builtin_amdgcn_wmma_f32_16x16x32_f16(false, ak0.v, false, bq0.v,
                                              (short)0, sv, false, false);
  sv = __builtin_amdgcn_wmma_f32_16x16x32_f16(false, ak1.v, false, bq1.v,
                                              (short)0, sv, false, false);

  const v4f* ap = (const v4f*)aptr;
  const float4* sp = (const float4*)sptr;
  const v4f a0 = __builtin_nontemporal_load(ap);        // streaming adj: TH=NT
  const v4f a1 = __builtin_nontemporal_load(ap + 1);
  const float4 s0 = sp[0], s1 = sp[1];
  const float av[8]  = {a0.x, a0.y, a0.z, a0.w, a1.x, a1.y, a1.z, a1.w};
  const float sjv[8] = {s0.x, s0.y, s0.z, s0.w, s1.x, s1.y, s1.z, s1.w};

  PK8 pk;
#pragma unroll
  for (int e = 0; e < 8; ++e) {
    const float sc = fmaf(sv[e], gate8, fmaf(csi, sjv[e], av[e]));
    const float p  = (av[e] == 0.0f) ? 0.0f : __expf(sc);
    rs += p;
    pk.h[e] = (_Float16)p;
  }
  return pk.u4;
}

__global__ __launch_bounds__(128) void k_attn(
    const float* __restrict__ adj, const float* __restrict__ hIn,
    const float* __restrict__ sArr, const _Float16* __restrict__ q,
    const _Float16* __restrict__ k, const _Float16* __restrict__ mT,
    const float* __restrict__ scal, float* __restrict__ z) {
  __shared__ float redO[4][32][32];
  __shared__ float redRS[4][32];

  const int lane   = threadIdx.x & 31;
  const int wave   = threadIdx.x >> 5;
  const int li     = lane & 15;
  const bool hiH   = lane >= 16;
  const int joff   = hiH ? 8 : 0;
  const int i0     = blockIdx.x << 4;
  const int i      = i0 + li;
  const float gate8 = scal[0] * 0.125f;          // gate / sqrt(64)
  const float csi   = scal[1] * sArr[i];

  // q rows as B operand (K=d, N=i): per-lane d = 8h+0..7 and 16+8h+0..7 per operand
  V16H bq0, bq1;
  {
    const float4* qp = (const float4*)(q + (size_t)i * 64 + (hiH ? 8 : 0));
    bq0.f4[0] = qp[0];
    bq0.f4[1] = qp[2];
    bq1.f4[0] = qp[4];
    bq1.f4[1] = qp[6];
  }

  v8f accO[4];
#pragma unroll
  for (int t = 0; t < 4; ++t)
#pragma unroll
    for (int e = 0; e < 8; ++e) accO[t][e] = 0.0f;
  float rs = 0.0f;

  const float* adj_row = adj + (size_t)i * NN;

  for (int j0 = wave * 32; j0 < NN; j0 += 128) {
    PK8 p0, p1;
    p0.u4 = score_tile(k + (size_t)(j0 + li) * 64 + joff, bq0, bq1,
                       adj_row + j0 + joff, sArr + j0 + joff, gate8, csi, rs);
    p1.u4 = score_tile(k + (size_t)(j0 + 16 + li) * 64 + joff, bq0, bq1,
                       adj_row + j0 + 16 + joff, sArr + j0 + 16 + joff,
                       gate8, csi, rs);

    PK8 sw0, sw1;
#pragma unroll
    for (int w = 0; w < 4; ++w) {
      sw0.u[w] = (unsigned)__builtin_amdgcn_ds_swizzle((int)p0.u[w], 0x401f);
      sw1.u[w] = (unsigned)__builtin_amdgcn_ds_swizzle((int)p1.u[w], 0x401f);
    }
    // B operand (K=j0..j0+31, N=i):
    //  low lanes : v0..3 = tile0 self (j 0..7), v4..7 = tile0 partner (j 8..15)
    //  high lanes: v0..3 = tile1 partner (j16..23), v4..7 = tile1 self (24..31)
    V16H bp;
#pragma unroll
    for (int w = 0; w < 4; ++w) {
      bp.u[w]     = hiH ? sw1.u[w] : p0.u[w];
      bp.u[w + 4] = hiH ? p1.u[w] : sw0.u[w];
    }

#pragma unroll
    for (int t = 0; t < 4; ++t) {
      V16H am;  // A = mT tile, M = d = 16t + li, K = j
      const float4* mp =
          (const float4*)(mT + (size_t)(t * 16 + li) * NN + j0 + (hiH ? 16 : 0));
      am.f4[0] = mp[0];
      am.f4[1] = mp[1];
      accO[t] = __builtin_amdgcn_wmma_f32_16x16x32_f16(false, am.v, false, bp.v,
                                                       (short)0, accO[t],
                                                       false, false);
    }
  }

  // cross-wave reduction of partial O^T and rowsum
  redRS[wave][lane] = rs;
#pragma unroll
  for (int t = 0; t < 4; ++t)
#pragma unroll
    for (int e = 0; e < 8; ++e) redO[wave][t * 8 + e][lane] = accO[t][e];
  __syncthreads();

  float rstot = 0.0f;
#pragma unroll
  for (int w = 0; w < 4; ++w) rstot += redRS[w][li] + redRS[w][li + 16];
  const float inv = 1.0f / rstot;   // diagonal edge guarantees rstot > 0

  const int t  = wave;              // this wave finalizes d-tile 16t..16t+15
  const int d0 = t * 16 + (hiH ? 8 : 0);
  const float* hp = hIn + (size_t)i * 64 + d0;
  float* zp       = z + (size_t)i * 64 + d0;
#pragma unroll
  for (int e = 0; e < 8; ++e) {
    const float v = redO[0][t * 8 + e][lane] + redO[1][t * 8 + e][lane] +
                    redO[2][t * 8 + e][lane] + redO[3][t * 8 + e][lane];
    zp[e] = fmaf(v, inv, hp[e]);
  }
}

// ---------------------------------------------------------------------------
// Phase 3: layernorm + actor head.  One 64-thread block per row.
// ---------------------------------------------------------------------------
__global__ __launch_bounds__(64) void k_head(
    const float* __restrict__ z, const float* __restrict__ sa,
    const float* __restrict__ lng, const float* __restrict__ lnb,
    const float* __restrict__ wa1, const float* __restrict__ ba1,
    const float* __restrict__ wa2, const float* __restrict__ ba2,
    float* __restrict__ out) {
  __shared__ float red[64];
  __shared__ float ai[66];
  __shared__ float hid[64];
  const int i = blockIdx.x;
  const int t = threadIdx.x;

  const float zv = z[(size_t)i * 64 + t];
  red[t] = zv;
  __syncthreads();
  for (int off = 32; off > 0; off >>= 1) {
    if (t < off) red[t] += red[t + off];
    __syncthreads();
  }
  const float mu = red[0] * (1.0f / 64.0f);
  __syncthreads();
  const float dz = zv - mu;
  red[t] = dz * dz;
  __syncthreads();
  for (int off = 32; off > 0; off >>= 1) {
    if (t < off) red[t] += red[t + off];
    __syncthreads();
  }
  const float var = red[0] * (1.0f / 64.0f);

  ai[t] = fmaf(dz * rsqrtf(var + 1e-5f), lng[t], lnb[t]);
  if (t < 2) ai[64 + t] = sa[(size_t)i * 2 + t];
  __syncthreads();

  float acc = ba1[t];
#pragma unroll 2
  for (int c = 0; c < 66; ++c) acc = fmaf(ai[c], wa1[c * 64 + t], acc);
  hid[t] = fmaxf(acc, 0.0f);
  __syncthreads();

  if (t < 3) {
    float o = ba2[t];
#pragma unroll 4
    for (int c = 0; c < 64; ++c) o = fmaf(hid[c], wa2[c * 3 + t], o);
    out[(size_t)i * 3 + t] = o;
  }
}

// ---------------------------------------------------------------------------
extern "C" void kernel_launch(void* const* d_in, const int* in_sizes, int n_in,
                              void* d_out, int out_size, void* d_ws,
                              size_t ws_size, hipStream_t stream) {
  (void)in_sizes; (void)n_in; (void)out_size; (void)ws_size;
  const float* nf   = (const float*)d_in[0];
  const float* adj  = (const float*)d_in[1];
  const float* trig = (const float*)d_in[2];
  const float* sa   = (const float*)d_in[3];
  const float* w1   = (const float*)d_in[4];
  const float* b1   = (const float*)d_in[5];
  const float* w2   = (const float*)d_in[6];
  const float* b2   = (const float*)d_in[7];
  const float* wq   = (const float*)d_in[8];
  const float* bq   = (const float*)d_in[9];
  const float* wk   = (const float*)d_in[10];
  const float* bk   = (const float*)d_in[11];
  const float* wg   = (const float*)d_in[12];
  const float* bg   = (const float*)d_in[13];
  const float* sb   = (const float*)d_in[14];
  const float* wm   = (const float*)d_in[15];
  const float* bm   = (const float*)d_in[16];
  const float* lng  = (const float*)d_in[17];
  const float* lnb  = (const float*)d_in[18];
  const float* wa1  = (const float*)d_in[19];
  const float* ba1  = (const float*)d_in[20];
  const float* wa2  = (const float*)d_in[21];
  const float* ba2  = (const float*)d_in[22];
  float* out = (float*)d_out;

  // workspace layout (bytes), all offsets 1 KB aligned; total ~7.04 MB
  char* ws = (char*)d_ws;
  float*    scal  = (float*)(ws + 0);            // [gate, c]
  float*    sArr  = (float*)(ws + 1024);         // 8192 f32
  float*    hBuf  = (float*)(ws + 33792);        // 8192x64 f32
  float*    zBuf  = (float*)(ws + 2130944);      // 8192x64 f32
  _Float16* qBuf  = (_Float16*)(ws + 4228096);   // 8192x64 f16
  _Float16* kBuf  = (_Float16*)(ws + 5276672);   // 8192x64 f16
  _Float16* mTBuf = (_Float16*)(ws + 6325248);   // 64x8192 f16 (transposed m)

  k_scalars<<<1, 64, 0, stream>>>(trig, wg, bg, sb, scal);
  k_embed<<<64, 256, 0, stream>>>(nf, trig, w1, b1, w2, b2, wq, bq, wk, bk, wm,
                                  bm, hBuf, qBuf, kBuf, mTBuf, sArr);
  k_attn<<<512, 128, 0, stream>>>(adj, hBuf, sArr, qBuf, kBuf, mTBuf, scal,
                                  zBuf);
  k_head<<<8192, 64, 0, stream>>>(zBuf, sa, lng, lnb, wa1, ba1, wa2, ba2, out);
}
